// MPXGAT_H_15831249453670
// MI455X (gfx1250) — matrix operand order repaired and checked by reading, not compiled, not run on hardware
//
#include <hip/hip_runtime.h>
#include <hip/hip_bf16.h>

// Problem constants (match reference)
#define GN   50000          // nodes
#define GE   800000         // edges (before self loops)
#define GH   3              // heads
#define CIN  128            // layer-1 input dim
#define CH   64             // channels per head
#define COUT 192            // GH*CH = GEMM output cols for every layer
#define NEG_SLOPE 0.2f

typedef __attribute__((ext_vector_type(16))) _Float16 v16h;
typedef __attribute__((ext_vector_type(8)))  float    v8f;

union HV { uint4 u4[2]; v16h h; };

// ---------------------------------------------------------------------------
// f32 -> f16 conversion (inputs / weights)
__global__ void gat_cvt_f32_f16(const float* __restrict__ in,
                                _Float16* __restrict__ out, int n) {
  int i = blockIdx.x * blockDim.x + threadIdx.x;
  if (i < n) out[i] = (_Float16)in[i];
}

// ---------------------------------------------------------------------------
// WMMA GEMM:  Y[M, 192] = Xh[M, K] * Wh[192, K]^T   (f16 in, f32 accumulate)
// One wave per 16x16 output tile. grid = (M/16, 192/16/4), block = 128 (4 waves).
__global__ void __launch_bounds__(128) gat_gemm_wmma(
    const _Float16* __restrict__ Xh, const _Float16* __restrict__ Wh,
    float* __restrict__ Y, int K) {
  const int lane = threadIdx.x & 31;
  const int wave = threadIdx.x >> 5;
  const int l16  = lane & 15;
  const int hi   = lane >> 4;                 // half-wave select
  const int mt   = blockIdx.x;                // 0..M/16-1
  const int nt   = blockIdx.y * 4 + wave;     // 0..11
  const int mrow = mt * 16 + l16;
  const int ncol = nt * 16 + l16;

  const unsigned int* Arow = (const unsigned int*)(Xh + (size_t)mrow * K);
  const unsigned int* Brow = (const unsigned int*)(Wh + (size_t)ncol * K);

  v8f acc = {};
  for (int k0 = 0; k0 < K; k0 += 32) {
    const unsigned int* a32 = Arow + (k0 >> 1);
    const unsigned int* b32 = Brow + (k0 >> 1);
    HV a, b;
    // A 16x32 f16 layout: VGPR v holds dword (v<4 ? v : v+4) + 4*hi of the row
    a.u4[0] = *(const uint4*)(a32 + 4 * hi);          // VGPRs 0-3
    a.u4[1] = *(const uint4*)(a32 + 8 + 4 * hi);      // VGPRs 4-7
    // B 32x16 f16 layout: VGPR v holds dword (v<4 ? v : v+4) + 4*hi of W's row `ncol`
    b.u4[0] = *(const uint4*)(b32 + 4 * hi);          // VGPRs 0-3
    b.u4[1] = *(const uint4*)(b32 + 8 + 4 * hi);      // VGPRs 4-7
    acc = __builtin_amdgcn_wmma_f32_16x16x32_f16(false, a.h, false, b.h,
                                                 (short)0, acc, false, false);
  }
  // D layout: VGPR r -> row mt*16 + r + 8*hi, col ncol
  float* yrow = Y + (size_t)(mt * 16 + 8 * hi) * COUT + ncol;
#pragma unroll
  for (int r = 0; r < 8; ++r)
    yrow[(size_t)r * COUT] = acc[r];
}

// ---------------------------------------------------------------------------
// Attention scores: s_src/s_dst [N, H]
__global__ void gat_scores(const float* __restrict__ xw,
                           const float* __restrict__ a_src,
                           const float* __restrict__ a_dst,
                           float* __restrict__ ssrc, float* __restrict__ sdst,
                           int n /* N*H */) {
  int i = blockIdx.x * blockDim.x + threadIdx.x;
  if (i >= n) return;
  int node = i / GH, h = i - node * GH;
  const float* row = xw + (size_t)node * COUT + h * CH;
  const float* as = a_src + h * CH;
  const float* ad = a_dst + h * CH;
  float vs = 0.f, vd = 0.f;
#pragma unroll 8
  for (int c = 0; c < CH; ++c) {
    float v = row[c];
    vs += v * as[c];
    vd += v * ad[c];
  }
  ssrc[i] = vs;
  sdst[i] = vd;
}

// ---------------------------------------------------------------------------
// Order-preserving float<->uint key for atomic max on floats
__device__ __forceinline__ unsigned int f2key(float f) {
  unsigned int b = __float_as_uint(f);
  return (b & 0x80000000u) ? ~b : (b | 0x80000000u);
}
__device__ __forceinline__ float key2f(unsigned int k) {
  unsigned int b = (k & 0x80000000u) ? (k ^ 0x80000000u) : ~k;
  return __uint_as_float(b);
}
__device__ __forceinline__ float lrelu(float x) {
  return x > 0.f ? x : NEG_SLOPE * x;
}
__device__ __forceinline__ float elu(float x) {
  return x > 0.f ? x : expm1f(x);
}
__device__ __forceinline__ void edge_sd(const int* __restrict__ ei, int i,
                                        int& s, int& d) {
  if (i < GE) { s = ei[i]; d = ei[GE + i]; }
  else        { s = i - GE; d = s; }        // self loop
}

// Pass 1: per-(dst,head) running max of leaky_relu(s_src[src]+s_dst[dst])
__global__ void gat_edge_max(const int* __restrict__ ei,
                             const float* __restrict__ ssrc,
                             const float* __restrict__ sdst,
                             unsigned int* __restrict__ mkey) {
  int idx = blockIdx.x * blockDim.x + threadIdx.x;
  const int tot = (GE + GN) * GH;
  if (idx >= tot) return;
  int i = idx / GH, h = idx - i * GH;
  int s, d; edge_sd(ei, i, s, d);
  float e = lrelu(ssrc[s * GH + h] + sdst[d * GH + h]);
  atomicMax(&mkey[d * GH + h], f2key(e));
}

// Pass 2: softmax denominator
__global__ void gat_edge_denom(const int* __restrict__ ei,
                               const float* __restrict__ ssrc,
                               const float* __restrict__ sdst,
                               const unsigned int* __restrict__ mkey,
                               float* __restrict__ den) {
  int idx = blockIdx.x * blockDim.x + threadIdx.x;
  const int tot = (GE + GN) * GH;
  if (idx >= tot) return;
  int i = idx / GH, h = idx - i * GH;
  int s, d; edge_sd(ei, i, s, d);
  float e = lrelu(ssrc[s * GH + h] + sdst[d * GH + h]);
  float m = key2f(mkey[d * GH + h]);
  atomicAdd(&den[d * GH + h], expf(e - m));
}

// Pass 3: weighted aggregation. One wave per edge, 192 channels (6 per lane).
__global__ void __launch_bounds__(256) gat_edge_aggregate(
    const int* __restrict__ ei, const float* __restrict__ ssrc,
    const float* __restrict__ sdst, const unsigned int* __restrict__ mkey,
    const float* __restrict__ den, const float* __restrict__ xw,
    float* __restrict__ agg) {
  const int wid  = blockIdx.x * (blockDim.x >> 5) + (threadIdx.x >> 5);
  const int lane = threadIdx.x & 31;
  const int EP = GE + GN;
  if (wid >= EP) return;
  int s, d; edge_sd(ei, wid, s, d);
  const float* xrow = xw + (size_t)s * COUT;
  float* orow = agg + (size_t)d * COUT;
  __builtin_prefetch(xrow + lane * 4, 0, 1);   // pull source row toward WGP
#pragma unroll
  for (int h = 0; h < GH; ++h) {
    float e = lrelu(ssrc[s * GH + h] + sdst[d * GH + h]);
    float m = key2f(mkey[d * GH + h]);
    float alpha = expf(e - m) / (den[d * GH + h] + 1e-16f);
    int c0 = h * CH + lane;
    atomicAdd(&orow[c0],      xrow[c0]      * alpha);
    atomicAdd(&orow[c0 + 32], xrow[c0 + 32] * alpha);
  }
}

// ---------------------------------------------------------------------------
// Finalize layers 1-2: y = elu(agg + b) -> f16 (next layer's GEMM input)
__global__ void gat_finalize_mid(const float* __restrict__ agg,
                                 const float* __restrict__ bias,
                                 _Float16* __restrict__ Xh, int n) {
  int i = blockIdx.x * blockDim.x + threadIdx.x;
  if (i >= n) return;
  int j = i % COUT;
  Xh[i] = (_Float16)elu(agg[i] + bias[j]);
}

// Finalize layer 3: z = elu(mean_heads(agg) + b) -> f32 output
__global__ void gat_finalize_last(const float* __restrict__ agg,
                                  const float* __restrict__ bias,
                                  float* __restrict__ z, int n) {
  int i = blockIdx.x * blockDim.x + threadIdx.x;
  if (i >= n) return;
  int node = i >> 6, c = i & 63;
  const float* row = agg + (size_t)node * COUT;
  float v = (row[c] + row[CH + c] + row[2 * CH + c]) * (1.f / 3.f) + bias[c];
  z[i] = elu(v);
}

// ---------------------------------------------------------------------------
static inline size_t alup(size_t x) { return (x + 255) & ~(size_t)255; }

extern "C" void kernel_launch(void* const* d_in, const int* in_sizes, int n_in,
                              void* d_out, int out_size, void* d_ws, size_t ws_size,
                              hipStream_t stream) {
  (void)in_sizes; (void)n_in; (void)out_size; (void)ws_size;
  // Workspace carve-up (~98 MB), reused for both graphs (run sequentially)
  char* p = (char*)d_ws;
  _Float16* Xh = (_Float16*)p;      p += alup((size_t)GN * COUT * 2);
  _Float16* Wh = (_Float16*)p;      p += alup((size_t)COUT * COUT * 2);
  float* xw    = (float*)p;         p += alup((size_t)GN * COUT * 4);
  float* agg   = (float*)p;         p += alup((size_t)GN * COUT * 4);
  float* ssrc  = (float*)p;         p += alup((size_t)GN * GH * 4);
  float* sdst  = (float*)p;         p += alup((size_t)GN * GH * 4);
  unsigned int* mkey = (unsigned int*)p; p += alup((size_t)GN * GH * 4);
  float* den   = (float*)p;         p += alup((size_t)GN * GH * 4);

  const int EP = GE + GN;
  // Input order: x0, x1, edge_index0, edge_index1, then params flattened
  // (pytree: per graph, per conv, keys sorted: W, a_src, a_dst, b)
  for (int g = 0; g < 2; ++g) {
    const float* x  = (const float*)d_in[g];
    const int*   ei = (const int*)d_in[2 + g];
    const int pb = 4 + g * 12;
    float* zout = (float*)d_out + (size_t)g * GN * CH;

    { int n = GN * CIN;
      gat_cvt_f32_f16<<<(n + 255) / 256, 256, 0, stream>>>(x, Xh, n); }

    for (int l = 0; l < 3; ++l) {
      const float* W    = (const float*)d_in[pb + l * 4 + 0];
      const float* a_s  = (const float*)d_in[pb + l * 4 + 1];
      const float* a_d  = (const float*)d_in[pb + l * 4 + 2];
      const float* bias = (const float*)d_in[pb + l * 4 + 3];
      const int K = (l == 0) ? CIN : COUT;

      { int n = COUT * K;
        gat_cvt_f32_f16<<<(n + 255) / 256, 256, 0, stream>>>(W, Wh, n); }

      gat_gemm_wmma<<<dim3(GN / 16, COUT / 16 / 4), 128, 0, stream>>>(Xh, Wh, xw, K);

      { int n = GN * GH;
        gat_scores<<<(n + 255) / 256, 256, 0, stream>>>(xw, a_s, a_d, ssrc, sdst, n); }

      hipMemsetAsync(mkey, 0, (size_t)GN * GH * 4, stream);
      hipMemsetAsync(den,  0, (size_t)GN * GH * 4, stream);
      hipMemsetAsync(agg,  0, (size_t)GN * COUT * 4, stream);

      { int n = EP * GH; int nb = (n + 255) / 256;
        gat_edge_max  <<<nb, 256, 0, stream>>>(ei, ssrc, sdst, mkey);
        gat_edge_denom<<<nb, 256, 0, stream>>>(ei, ssrc, sdst, mkey, den); }

      { int nb = (EP + 7) / 8;   // 8 waves per block, wave per edge
        gat_edge_aggregate<<<nb, 256, 0, stream>>>(ei, ssrc, sdst, mkey, den, xw, agg); }

      if (l < 2) {
        int n = GN * COUT;
        gat_finalize_mid<<<(n + 255) / 256, 256, 0, stream>>>(agg, bias, Xh, n);
      } else {
        int n = GN * CH;
        gat_finalize_last<<<(n + 255) / 256, 256, 0, stream>>>(agg, bias, zout, n);
      }
    }
  }
}
